// CrossAttention_13469017440539
// MI455X (gfx1250) — matrix operand order repaired and checked by reading, not compiled, not run on hardware
//
#include <hip/hip_runtime.h>
#include <stdint.h>

// Problem constants (from reference): B=8, N1=N2=2048, D=120 (pad to 128)
#define B_   8
#define N_   2048
#define D_   120
#define DP_  128
#define SCALE_ 0.09128709291752768f   // 1/sqrt(120), folded into q

typedef __attribute__((ext_vector_type(16))) __bf16         v16bf;
typedef __attribute__((ext_vector_type(8)))  float          v8f;
typedef __attribute__((ext_vector_type(16))) unsigned short v16u;
typedef __attribute__((ext_vector_type(8)))  unsigned short v8u;
typedef __attribute__((ext_vector_type(4)))  float          v4f;

static __device__ __forceinline__ unsigned short f2bf(float f) {
  unsigned int u = __float_as_uint(f);
  u += 0x7fffu + ((u >> 16) & 1u);          // round-to-nearest-even
  return (unsigned short)(u >> 16);
}

// ---------------------------------------------------------------------------
// Kernel 1: QKV projection, fp32 dot -> bf16, D padded 120->128 with zeros.
// z=0: q = (x1 @ Wq) * scale  -> qb  [b][n][128]
// z=1: k =  x2 @ Wk           -> kb  [b][n][128]
// z=2: v =  x2 @ Wv           -> vtb [b][d=128][n]   (pre-transposed for PV B-frags)
// ---------------------------------------------------------------------------
__global__ __launch_bounds__(256) void ca_proj(
    const float* __restrict__ x1, const float* __restrict__ x2,
    const float* __restrict__ Wq, const float* __restrict__ Wk,
    const float* __restrict__ Wv,
    unsigned short* __restrict__ qb, unsigned short* __restrict__ kb,
    unsigned short* __restrict__ vtb)
{
  const int tid = blockIdx.x * 256 + threadIdx.x;   // 0 .. N_*DP_-1
  const int n = tid >> 7;
  const int e = tid & 127;
  const int b = blockIdx.y;
  const int w = blockIdx.z;
  const float* x = (w == 0) ? x1 : x2;
  const float* W = (w == 0) ? Wq : (w == 1) ? Wk : Wv;
  float acc = 0.f;
  if (e < D_) {
    const float* xr = x + ((size_t)b * N_ + n) * D_;
    #pragma unroll 8
    for (int d = 0; d < D_; ++d) acc += xr[d] * W[d * D_ + e];
  }
  if (w == 0) acc *= SCALE_;
  const unsigned short h = f2bf(acc);
  if (w == 0)      qb [((size_t)b * N_  + n) * DP_ + e] = h;
  else if (w == 1) kb [((size_t)b * N_  + n) * DP_ + e] = h;
  else             vtb[((size_t)b * DP_ + e) * N_  + n] = h;
}

// ---------------------------------------------------------------------------
// Kernel 2: attention. 128 threads = 4 wave32; each wave owns one 16-row tile.
// Phase 1: raw scores via v_wmma_f32_16x16x32_bf16 (K=128 in 4 chunks),
//          mask to -inf, write raw scores, track row max in registers.
// Phase 2: exp-sum per row (raw scores re-read from L2-resident buffer).
// Phase 3: normalized P written in place + O = P @ V via WMMA (V^T layout).
// ---------------------------------------------------------------------------
__global__ __launch_bounds__(128) void ca_attn(
    const unsigned short* __restrict__ qb, const unsigned short* __restrict__ kb,
    const unsigned short* __restrict__ vtb, const int* __restrict__ mask,
    float* __restrict__ outO, float* __restrict__ attn)
{
  __shared__ float sm_max[4][16];
  __shared__ float sm_rl [4][16];

  const int b    = blockIdx.y;
  const int wave = threadIdx.x >> 5;
  const int lane = threadIdx.x & 31;
  const int m16  = lane & 15;       // A: row / B: col / C: col
  const int half = lane >> 4;       // lane-group select per ISA layouts
  const int mbase = (blockIdx.x * 4 + wave) * 16;

  // ---- Q A-fragments: 16x32 bf16 per chunk; per-lane = two 16B runs:
  // element i<8 : k = half*8 + i ; element i>=8 : k = 16 + half*8 + (i-8)
  const unsigned short* qrow = qb + ((size_t)b * N_ + mbase + m16) * DP_;
  v16u aq[4];
  #pragma unroll
  for (int c = 0; c < 4; ++c) {
    v8u lo = *(const v8u*)(qrow + c * 32 + half * 8);
    v8u hi = *(const v8u*)(qrow + c * 32 + 16 + half * 8);
    aq[c] = __builtin_shufflevector(lo, hi, 0,1,2,3,4,5,6,7,8,9,10,11,12,13,14,15);
  }

  const int*  maskrow = mask + ((size_t)b * N_ + mbase) * N_;
  float*      attnrow = attn + ((size_t)b * N_ + mbase) * N_;

  float mrun[8];
  #pragma unroll
  for (int r = 0; r < 8; ++r) mrun[r] = -__builtin_inff();

  // ---- Phase 1: scores ----
  for (int j = 0; j < N_ / 16; ++j) {
    const int cbase = j * 16;
    v8f acc = {};
    // B-frag of K^T: element i<8 : k = half*8 + i ; i>=8 : k = 16 + half*8 + (i-8), n = m16
    // -> per lane two 16B runs from kb row (cbase+m16)
    const unsigned short* krow =
        kb + ((size_t)b * N_ + cbase + m16) * DP_ + half * 8;
    #pragma unroll
    for (int c = 0; c < 4; ++c) {
      v8u l0 = *(const v8u*)(krow + c * 32);
      v8u l1 = *(const v8u*)(krow + c * 32 + 16);
      v16u bk = __builtin_shufflevector(l0, l1, 0,1,2,3,4,5,6,7,8,9,10,11,12,13,14,15);
      acc = __builtin_amdgcn_wmma_f32_16x16x32_bf16(
          false, __builtin_bit_cast(v16bf, aq[c]),
          false, __builtin_bit_cast(v16bf, bk),
          (short)0, acc, false, false);
    }
    // C layout: VGPR r -> row r + 8*half, col = m16
    #pragma unroll
    for (int r = 0; r < 8; ++r) {
      const int row = r + 8 * half;
      float s = acc[r];
      const int mk = maskrow[(size_t)row * N_ + cbase + m16];
      s = mk ? s : -__builtin_inff();
      mrun[r] = fmaxf(mrun[r], s);
      attnrow[(size_t)row * N_ + cbase + m16] = s;     // raw scaled score
    }
  }
  // row-max reduce across the 16 lanes of each half-group (xor stays in group)
  #pragma unroll
  for (int r = 0; r < 8; ++r) {
    #pragma unroll
    for (int off = 1; off < 16; off <<= 1)
      mrun[r] = fmaxf(mrun[r], __shfl_xor(mrun[r], off, 32));
  }
  if (m16 == 0) {
    #pragma unroll
    for (int r = 0; r < 8; ++r) sm_max[wave][half * 8 + r] = mrun[r];
  }
  __syncthreads();

  // ---- Phase 2: exp-sum per row (2 lanes per row, 1024 elems each) ----
  {
    const int   row  = lane >> 1;
    const int   coff = lane & 1;
    const float mx   = sm_max[wave][row];
    const float* ar  = attnrow + (size_t)row * N_;
    float sum = 0.f;
    for (int t = 0; t < N_ / 2; ++t) sum += __expf(ar[coff + 2 * t] - mx);
    sum += __shfl_xor(sum, 1, 32);
    if (coff == 0) sm_rl[wave][row] = 1.0f / sum;
  }
  __syncthreads();

  // ---- Phase 3: normalized weights (written back) + O = P @ V ----
  v8f oacc[8];
  #pragma unroll
  for (int f = 0; f < 8; ++f) oacc[f] = (v8f){};

  const float rm = sm_max[wave][m16];   // A-frag row = m16
  const float rl = sm_rl [wave][m16];
  float* prow = attnrow + (size_t)m16 * N_;

  for (int jc = 0; jc < N_ / 32; ++jc) {
    const int cb = jc * 32;
    float* r0 = prow + cb + half * 8;        // k = half*8 + 0..7
    float* r1 = prow + cb + 16 + half * 8;   // k = 16 + half*8 + 0..7
    v4f a0 = *(const v4f*)(r0);
    v4f a1 = *(const v4f*)(r0 + 4);
    v4f b0 = *(const v4f*)(r1);
    v4f b1 = *(const v4f*)(r1 + 4);
    float p[16];
    #pragma unroll
    for (int i = 0; i < 4; ++i) {
      p[i]      = __expf(a0[i] - rm) * rl;
      p[4 + i]  = __expf(a1[i] - rm) * rl;
      p[8 + i]  = __expf(b0[i] - rm) * rl;
      p[12 + i] = __expf(b1[i] - rm) * rl;
    }
    // overwrite raw scores with normalized attention weights (tuple output #2)
    v4f w0, w1, w2, w3;
    #pragma unroll
    for (int i = 0; i < 4; ++i) { w0[i] = p[i]; w1[i] = p[4+i]; w2[i] = p[8+i]; w3[i] = p[12+i]; }
    *(v4f*)(r0)     = w0;  *(v4f*)(r0 + 4) = w1;
    *(v4f*)(r1)     = w2;  *(v4f*)(r1 + 4) = w3;
    // bf16 A-fragment of P
    v16u ap;
    #pragma unroll
    for (int i = 0; i < 16; ++i) ap[i] = f2bf(p[i]);
    // B-frags of V from V^T: per lane, d = f*16 + m16 fixed, two 16B runs
    // along n at cb + half*8 and cb + 16 + half*8
    #pragma unroll
    for (int f = 0; f < 8; ++f) {
      const unsigned short* vr =
          vtb + ((size_t)b * DP_ + f * 16 + m16) * N_ + cb + half * 8;
      v8u l0 = *(const v8u*)(vr);
      v8u l1 = *(const v8u*)(vr + 16);
      v16u bv = __builtin_shufflevector(l0, l1, 0,1,2,3,4,5,6,7,8,9,10,11,12,13,14,15);
      oacc[f] = __builtin_amdgcn_wmma_f32_16x16x32_bf16(
          false, __builtin_bit_cast(v16bf, ap),
          false, __builtin_bit_cast(v16bf, bv),
          (short)0, oacc[f], false, false);
    }
  }

  // ---- Phase 4: store O (drop padded d=120..127) ----
  float* orow = outO + ((size_t)b * N_ + mbase) * D_;
  #pragma unroll
  for (int f = 0; f < 8; ++f) {
    const int d = f * 16 + m16;
    if (d < D_) {
      #pragma unroll
      for (int r = 0; r < 8; ++r)
        orow[(size_t)(r + 8 * half) * D_ + d] = oacc[f][r];
    }
  }
}

// ---------------------------------------------------------------------------
extern "C" void kernel_launch(void* const* d_in, const int* in_sizes, int n_in,
                              void* d_out, int out_size, void* d_ws, size_t ws_size,
                              hipStream_t stream) {
  (void)in_sizes; (void)n_in; (void)out_size; (void)ws_size;
  const float* x1   = (const float*)d_in[0];
  const float* x2   = (const float*)d_in[1];
  const int*   mask = (const int*)  d_in[2];
  const float* Wq   = (const float*)d_in[3];
  const float* Wk   = (const float*)d_in[4];
  const float* Wv   = (const float*)d_in[5];

  float* out  = (float*)d_out;                       // [B,N1,120]
  float* attn = out + (size_t)B_ * N_ * D_;          // [B,N1,N2]

  unsigned short* qb  = (unsigned short*)d_ws;               // 4 MB
  unsigned short* kb  = qb + (size_t)B_ * N_ * DP_;          // 4 MB
  unsigned short* vtb = kb + (size_t)B_ * N_ * DP_;          // 4 MB (V^T)

  dim3 gp((N_ * DP_) / 256, B_, 3);
  ca_proj<<<gp, 256, 0, stream>>>(x1, x2, Wq, Wk, Wv, qb, kb, vtb);

  dim3 ga(N_ / 64, B_);   // 4 waves/block, 16 rows per wave
  ca_attn<<<ga, 128, 0, stream>>>(qb, kb, vtb, mask, out, attn);
}
